// Histoformer_13872744366681
// MI455X (gfx1250) — compile-verified, matrix operand order checked by reading, not run on hardware
//
#include <hip/hip_runtime.h>
#include <hip/hip_bf16.h>

// ---------------- problem constants ----------------
#define B_     4
#define C_     64
#define HALF_  32
#define H_     256
#define W_     256
#define HW_    65536ull        // H_*W_
#define C5_    320             // 5*C
#define HEADS_ 8
#define NMAT_  64              // 2 branches * B * HEADS

typedef __attribute__((ext_vector_type(16))) _Float16 v16h;
typedef __attribute__((ext_vector_type(8)))  float    v8f;
typedef int v4i __attribute__((vector_size(16)));      // async-LDS builtin operand type

#if defined(__has_builtin)
# if __has_builtin(__builtin_amdgcn_global_load_async_to_lds_b128) && \
     __has_builtin(__builtin_amdgcn_s_wait_asynccnt)
#  define HAVE_ASYNC_LDS 1
# endif
#endif

// ---------------- helpers ----------------
__device__ __forceinline__ unsigned f2key(float f) {
    unsigned u = __float_as_uint(f);
    return (u & 0x80000000u) ? ~u : (u | 0x80000000u);
}
__device__ __forceinline__ float key2f(unsigned k) {
    unsigned u = (k & 0x80000000u) ? (k ^ 0x80000000u) : ~k;
    return __uint_as_float(u);
}

// stable 256-element bitonic sort in LDS; (value, original-index) total order
__device__ void bitonic256(float* sval, unsigned short* sidx) {
    int t = threadIdx.x;
    for (int k = 2; k <= 256; k <<= 1) {
        for (int j = k >> 1; j > 0; j >>= 1) {
            __syncthreads();
            int p = t ^ j;
            float va = sval[t], vb = sval[p];
            unsigned short ia = sidx[t], ib = sidx[p];
            bool up = ((t & k) == 0);
            bool mineGreater = (va > vb) || (va == vb && ia > ib);
            bool lower = (t < p);
            bool take = up ? (lower ? mineGreater : !mineGreater)
                           : (lower ? !mineGreater : mineGreater);
            __syncthreads();
            if (take) { sval[t] = vb; sidx[t] = ib; }
        }
    }
    __syncthreads();
}

// ---------------- kernels ----------------
__global__ void k_copy(const float* __restrict__ src, float* __restrict__ dst, size_t n) {
    size_t i = (size_t)blockIdx.x * blockDim.x + threadIdx.x;
    if (i < n) dst[i] = src[i];
}

__global__ void k_zero(float* __restrict__ p, size_t n) {
    size_t i = (size_t)blockIdx.x * blockDim.x + threadIdx.x;
    if (i < n) p[i] = 0.f;
}

__global__ void k_packw(const float* __restrict__ w, _Float16* __restrict__ wh, int n) {
    int i = blockIdx.x * blockDim.x + threadIdx.x;
    if (i < n) wh[i] = (_Float16)w[i];
}

// sort along H for channels < 32 (one column per block); record idx_h
__global__ void k_sort_h(float* __restrict__ xw, unsigned short* __restrict__ idxh) {
    __shared__ float sval[256];
    __shared__ unsigned short sidx[256];
    int w = blockIdx.x, c = blockIdx.y, b = blockIdx.z;
    int t = threadIdx.x;
    size_t base = ((size_t)(b * C_ + c)) << 16;
    sval[t] = xw[base + ((size_t)t << 8) + w];
    sidx[t] = (unsigned short)t;
    bitonic256(sval, sidx);
    xw[base + ((size_t)t << 8) + w] = sval[t];
    idxh[(((size_t)(b * HALF_ + c)) << 16) + ((size_t)t << 8) + w] = sidx[t];
}

// sort along W (one row per block); record idx_w
__global__ void k_sort_w(float* __restrict__ xw, unsigned short* __restrict__ idxw) {
    __shared__ float sval[256];
    __shared__ unsigned short sidx[256];
    int h = blockIdx.x, c = blockIdx.y, b = blockIdx.z;
    int t = threadIdx.x;
    size_t base = (((size_t)(b * C_ + c)) << 16) + ((size_t)h << 8);
    sval[t] = xw[base + t];
    sidx[t] = (unsigned short)t;
    bitonic256(sval, sidx);
    xw[base + t] = sval[t];
    idxw[(((size_t)(b * HALF_ + c)) << 16) + ((size_t)h << 8) + t] = sidx[t];
}

// transpose per-batch activations: X f32 [b][k=64][n] -> Xt f16 [b][n][k=64]
// grid(HW/64, B), block 256, 64x64 tile staged through LDS (conflict-free pad)
__global__ void k_transpose_f16(const float* __restrict__ X, _Float16* __restrict__ Xt) {
    __shared__ float tile[64][65];
    int b = blockIdx.y;
    int n0 = blockIdx.x * 64;
    const float* Xb = X + (size_t)b * 64 * HW_;
    int tx = threadIdx.x & 63;
    int ty = threadIdx.x >> 6;
    for (int k = ty; k < 64; k += 4)
        tile[k][tx] = Xb[((size_t)k << 16) + n0 + tx];
    __syncthreads();
    _Float16* Ot = Xt + ((size_t)b * HW_ + n0) * 64;
    for (int n = ty; n < 64; n += 4)
        Ot[(size_t)n * 64 + tx] = (_Float16)tile[tx][n];
}

// 1x1 conv as f16 WMMA GEMM: Y[b,m,n] = sum_k Wh[m,k] * Xt[b,n,k]  (K=64)
// grid(HW/64, M/16, B), block 128: wave owns one 16x16 tile; contiguous v16h loads
__global__ void k_conv1x1h(const _Float16* __restrict__ Wh, const _Float16* __restrict__ Xt,
                           float* __restrict__ Y, int M) {
    int wave = threadIdx.x >> 5, lane = threadIdx.x & 31;
    int lm = lane & 15, lh = lane >> 4;
    int colBase = blockIdx.x * 64 + wave * 16;
    int rowBase = blockIdx.y * 16;
    const _Float16* Xb = Xt + (size_t)blockIdx.z * HW_ * 64;
    float* Yb = Y + (size_t)blockIdx.z * M * HW_;
    v8f acc = {};
#pragma unroll
    for (int kk = 0; kk < 64; kk += 32) {
        v16h a = *(const v16h*)(Wh + (rowBase + lm) * 64 + kk + lh * 16);
        v16h b = *(const v16h*)(Xb + ((size_t)(colBase + lm) << 6) + kk + lh * 16);
        acc = __builtin_amdgcn_wmma_f32_16x16x32_f16(false, a, false, b,
                                                     (short)0, acc, false, false);
    }
#pragma unroll
    for (int j = 0; j < 8; ++j)
        Yb[((size_t)(rowBase + j + 8 * lh) << 16) + colBase + lm] = acc[j];
}

// depthwise 3x3, SAME zero pad
__global__ void k_dwconv(const float* __restrict__ in, const float* __restrict__ wdw,
                         float* __restrict__ out, size_t total) {
    size_t i = (size_t)blockIdx.x * blockDim.x + threadIdx.x;
    if (i >= total) return;
    int w = (int)(i & 255), h = (int)((i >> 8) & 255);
    size_t bc = i >> 16;
    int ch = (int)(bc % C5_);
    const float* wp = wdw + ch * 9;
    const float* src = in + (bc << 16);
    float s = 0.f;
#pragma unroll
    for (int dy = 0; dy < 3; ++dy) {
        int hh = h + dy - 1;
        if (hh < 0 || hh > 255) continue;
#pragma unroll
        for (int dx = 0; dx < 3; ++dx) {
            int ww = w + dx - 1;
            if (ww < 0 || ww > 255) continue;
            s += wp[dy * 3 + dx] * src[(hh << 8) + ww];
        }
    }
    out[i] = s;
}

// build sortable keys + indices for the per-channel argsort of v
__global__ void k_makekeys(const float* __restrict__ qkv_dw, unsigned* __restrict__ vkey,
                           unsigned* __restrict__ vidx, size_t total) {
    size_t i = (size_t)blockIdx.x * blockDim.x + threadIdx.x;
    if (i >= total) return;
    unsigned n = (unsigned)(i & 65535);
    size_t bc = i >> 16;
    size_t b = bc >> 6, c = bc & 63;
    float v = qkv_dw[((b * C5_ + 4 * C_ + c) << 16) + n];
    vkey[i] = f2key(v);
    vidx[i] = n;
}

// one global bitonic compare-exchange pass over each 65536-element channel
__global__ void k_bitonic(unsigned* __restrict__ key, unsigned* __restrict__ idx,
                          int j, int k, size_t total) {
    size_t i = (size_t)blockIdx.x * blockDim.x + threadIdx.x;
    if (i >= total) return;
    unsigned n = (unsigned)(i & 65535);
    if (n & (unsigned)j) return;
    size_t base = i - n;
    size_t A = base + n, P = base + (n ^ (unsigned)j);
    unsigned ka = key[A], kb = key[P];
    unsigned ia = idx[A], ib = idx[P];
    bool up = ((n & (unsigned)k) == 0);
    bool greater = (ka > kb) || (ka == kb && ia > ib);
    if (greater == up) {
        key[A] = kb; key[P] = ka;
        idx[A] = ib; idx[P] = ia;
    }
}

// pack gathered Q/K (row-major 64x8192 f16) and V transposed (8192x64 f16) per mat
__global__ void k_pack(const float* __restrict__ qkv_dw, const unsigned* __restrict__ vkey,
                       const unsigned* __restrict__ vidx,
                       _Float16* __restrict__ Qp, _Float16* __restrict__ Kp,
                       _Float16* __restrict__ Vp, size_t total) {
    size_t i = (size_t)blockIdx.x * blockDim.x + threadIdx.x;
    if (i >= total) return;
    int p = (int)(i & 8191);
    size_t t = i >> 13;
    int r = (int)(t & 63); t >>= 6;
    int h = (int)(t & 7);  t >>= 3;
    int b = (int)(t & 3);  t >>= 2;
    int br = (int)t;
    int chi = r >> 3, f = r & 7;
    int c = h * 8 + chi;
    int n = (br == 0) ? ((f << 13) | p) : ((p << 3) | f);
    size_t ch = ((size_t)b * C_ + c) << 16;
    unsigned src = vidx[ch + n];
    int qc = (br == 0 ? 0 : 2 * C_) + c;
    int kc = (br == 0 ? C_ : 3 * C_) + c;
    size_t bb = (size_t)b * C5_;
    float qv = qkv_dw[((bb + qc) << 16) + src];
    float kv = qkv_dw[((bb + kc) << 16) + src];
    float vv = key2f(vkey[ch + n]);
    size_t mat = (((size_t)br * B_ + b) * HEADS_ + h);
    size_t mb = mat << 19;
    Qp[mb + ((size_t)r << 13) + p] = (_Float16)qv;
    Kp[mb + ((size_t)r << 13) + p] = (_Float16)kv;
    Vp[mb + ((size_t)p << 6) + r] = (_Float16)vv;
}

// reciprocal L2 norms of the 8192-long rows of Qp (y==0) / Kp (y==1)
__global__ void k_rownorm(const _Float16* __restrict__ Qp, const _Float16* __restrict__ Kp,
                          float* __restrict__ rnq, float* __restrict__ rnk) {
    __shared__ float red[256];
    int row = blockIdx.x;
    const _Float16* src = (blockIdx.y == 0 ? Qp : Kp) + ((size_t)row << 13);
    float* dst = blockIdx.y == 0 ? rnq : rnk;
    float s = 0.f;
    for (int i = threadIdx.x; i < 8192; i += 256) {
        float v = (float)src[i];
        s += v * v;
    }
    red[threadIdx.x] = s; __syncthreads();
    for (int st = 128; st > 0; st >>= 1) {
        if (threadIdx.x < st) red[threadIdx.x] += red[threadIdx.x + st];
        __syncthreads();
    }
    if (threadIdx.x == 0) dst[row] = 1.0f / fmaxf(sqrtf(red[0]), 1e-12f);
}

// S = Q K^T (64x64, K-dim 8192) per mat; K split 8 ways with atomic reduction.
// K-tile (64 rows x 32 k, f16, 4KB) staged in LDS (shared by all 4 waves),
// via async global->LDS DMA when available.
__global__ void k_qkT(const _Float16* __restrict__ Qp, const _Float16* __restrict__ Kp,
                      float* __restrict__ S) {
    __shared__ __attribute__((aligned(32))) _Float16 kt[64 * 32];
    int mat = blockIdx.y;
    int tid = threadIdx.x;
    int wave = tid >> 5, lane = tid & 31;
    int lm = lane & 15, lh = lane >> 4;
    const _Float16* Q  = Qp + ((size_t)mat << 19);
    const _Float16* Km = Kp + ((size_t)mat << 19);
    float* Sb = S + ((size_t)mat << 12);
    int rowBase = wave * 16;
    int srow = tid >> 1, shalf = tid & 1;            // 128 threads stage 64x32 f16
    const _Float16* gsrc0 = Km + ((size_t)srow << 13) + shalf * 16;
    _Float16* ldst = kt + srow * 32 + shalf * 16;
    v8f a0 = {}, a1 = {}, a2 = {}, a3 = {};
    int k0 = blockIdx.x * 1024;
    for (int kk = k0; kk < k0 + 1024; kk += 32) {
        const _Float16* gsrc = gsrc0 + kk;
#ifdef HAVE_ASYNC_LDS
        __builtin_amdgcn_global_load_async_to_lds_b128(
            (__attribute__((address_space(1))) v4i*)gsrc,
            (__attribute__((address_space(3))) v4i*)ldst, 0, 0);
        __builtin_amdgcn_global_load_async_to_lds_b128(
            (__attribute__((address_space(1))) v4i*)(gsrc + 8),
            (__attribute__((address_space(3))) v4i*)(ldst + 8), 0, 0);
        __builtin_amdgcn_s_wait_asynccnt(0);
#else
        *(v16h*)ldst = *(const v16h*)gsrc;
#endif
        __syncthreads();
        const _Float16* qp = Q + ((size_t)(rowBase + lm) << 13) + kk + lh * 16;
        __builtin_prefetch(qp + 256, 0, 1);          // global_prefetch_b8
        v16h a = *(const v16h*)qp;
        v16h b0 = *(const v16h*)(kt + (lm)*32      + lh * 16);
        v16h b1 = *(const v16h*)(kt + (16 + lm)*32 + lh * 16);
        v16h b2 = *(const v16h*)(kt + (32 + lm)*32 + lh * 16);
        v16h b3 = *(const v16h*)(kt + (48 + lm)*32 + lh * 16);
        a0 = __builtin_amdgcn_wmma_f32_16x16x32_f16(false, a, false, b0, (short)0, a0, false, false);
        a1 = __builtin_amdgcn_wmma_f32_16x16x32_f16(false, a, false, b1, (short)0, a1, false, false);
        a2 = __builtin_amdgcn_wmma_f32_16x16x32_f16(false, a, false, b2, (short)0, a2, false, false);
        a3 = __builtin_amdgcn_wmma_f32_16x16x32_f16(false, a, false, b3, (short)0, a3, false, false);
        __syncthreads();
    }
#pragma unroll
    for (int j = 0; j < 8; ++j) {
        int r = rowBase + j + 8 * lh;
        atomicAdd(&Sb[r * 64 +      lm], a0[j]);
        atomicAdd(&Sb[r * 64 + 16 + lm], a1[j]);
        atomicAdd(&Sb[r * 64 + 32 + lm], a2[j]);
        atomicAdd(&Sb[r * 64 + 48 + lm], a3[j]);
    }
}

// scale logits by norms + temperature, softmax_1 along last dim, emit f16 attn
__global__ void k_softmax1(const float* __restrict__ S, const float* __restrict__ rnq,
                           const float* __restrict__ rnk, const float* __restrict__ temp,
                           _Float16* __restrict__ attnp) {
    __shared__ float red[64];
    __shared__ float ssum;
    int row = blockIdx.x & 63;
    int mat = blockIdx.x >> 6;
    int h = mat & 7;
    int s = threadIdx.x;
    float v = S[((size_t)mat << 12) + row * 64 + s]
            * rnq[mat * 64 + row] * rnk[mat * 64 + s] * temp[h];
    float e = expf(v);
    red[s] = e; __syncthreads();
    for (int st = 32; st > 0; st >>= 1) {
        if (s < st) red[s] += red[s + st];
        __syncthreads();
    }
    if (s == 0) ssum = red[0] + 1.0f;
    __syncthreads();
    attnp[((size_t)mat << 12) + row * 64 + s] = (_Float16)(e / ssum);
}

// out = attn(64x64) @ V(64x8192); grid(512, NMAT), block 128; un-reshape on store
__global__ void k_av(const _Float16* __restrict__ attnp, const _Float16* __restrict__ Vp,
                     float* __restrict__ out1, float* __restrict__ out2) {
    int mat = blockIdx.y;
    int wave = threadIdx.x >> 5, lane = threadIdx.x & 31;
    int lm = lane & 15, lh = lane >> 4;
    const _Float16* A  = attnp + ((size_t)mat << 12);
    const _Float16* Vm = Vp + ((size_t)mat << 19);
    int rowBase = wave * 16;
    int colBase = blockIdx.x * 16;
    v8f acc = {};
#pragma unroll
    for (int kk = 0; kk < 64; kk += 32) {
        v16h a = *(const v16h*)(A + (rowBase + lm) * 64 + kk + lh * 16);
        v16h b = *(const v16h*)(Vm + ((size_t)(colBase + lm) << 6) + kk + lh * 16);
        acc = __builtin_amdgcn_wmma_f32_16x16x32_f16(false, a, false, b,
                                                     (short)0, acc, false, false);
    }
    int br = mat >> 5, bh = mat & 31, b_ = bh >> 3, h = bh & 7;
    float* outb = (br == 0) ? out1 : out2;
#pragma unroll
    for (int j = 0; j < 8; ++j) {
        int r = rowBase + j + 8 * lh;
        int p = colBase + lm;
        int chi = r >> 3, f = r & 7;
        int c = h * 8 + chi;
        int n = (br == 0) ? ((f << 13) | p) : ((p << 3) | f);
        outb[(((size_t)b_ * C_ + c) << 16) + n] = acc[j];
    }
}

// product in sorted space scattered back through inverse sort permutation,
// written directly in the f16 transposed [b][n][c] layout the final GEMM wants
__global__ void k_mulscatter(const float* __restrict__ o1, const float* __restrict__ o2,
                             const unsigned* __restrict__ vidx, _Float16* __restrict__ prodt,
                             size_t total) {
    size_t i = (size_t)blockIdx.x * blockDim.x + threadIdx.x;
    if (i >= total) return;
    unsigned n = (unsigned)(i & 65535);
    size_t bc = i >> 16;
    int c = (int)(bc & 63);
    size_t b = bc >> 6;
    unsigned dst = vidx[i];
    prodt[(((size_t)b << 16) + dst) * 64 + c] = (_Float16)(o1[i] * o2[i]);
}

// copy conv output to d_out, un-sorting first 32 channels through idx_w then idx_h
__global__ void k_final(const float* __restrict__ cv, const unsigned short* __restrict__ idxh,
                        const unsigned short* __restrict__ idxw, float* __restrict__ out,
                        size_t total) {
    size_t i = (size_t)blockIdx.x * blockDim.x + threadIdx.x;
    if (i >= total) return;
    int wp = (int)(i & 255), hp = (int)((i >> 8) & 255);
    size_t bc = i >> 16;
    int c = (int)(bc & 63);
    int b = (int)(bc >> 6);
    if (c >= HALF_) { out[i] = cv[i]; return; }
    size_t hb = ((size_t)(b * HALF_ + c)) << 16;
    int ww = idxw[hb + ((size_t)hp << 8) + wp];
    int hh = idxh[hb + ((size_t)hp << 8) + ww];
    out[(bc << 16) + ((size_t)hh << 8) + ww] = cv[i];
}

// ---------------- host ----------------
extern "C" void kernel_launch(void* const* d_in, const int* in_sizes, int n_in,
                              void* d_out, int out_size, void* d_ws, size_t ws_size,
                              hipStream_t stream) {
    const float* x     = (const float*)d_in[0];
    const float* w_qkv = (const float*)d_in[1];
    const float* w_dw  = (const float*)d_in[2];
    const float* w_out = (const float*)d_in[3];
    const float* temp  = (const float*)d_in[4];

    const size_t NX    = (size_t)B_ * C_ * HW_;     // 16,777,216
    const size_t NQKV  = (size_t)B_ * C5_ * HW_;    // 83,886,080
    const size_t NPACK = (size_t)NMAT_ * 64 * 8192; // 33,554,432

    char* ws = (char*)d_ws;
    size_t off = 0;
    auto alloc = [&](size_t bytes) -> void* {
        void* p = ws + off;
        off += (bytes + 255) & ~(size_t)255;
        return p;
    };
    float*          x_work = (float*)alloc(NX * 4);
    unsigned short* idxh   = (unsigned short*)alloc((size_t)B_ * HALF_ * HW_ * 2);
    unsigned short* idxw   = (unsigned short*)alloc((size_t)B_ * HALF_ * HW_ * 2);
    float*          qkv    = (float*)alloc(NQKV * 4);     // reused post-dwconv
    float*          qkv_dw = (float*)alloc(NQKV * 4);     // reused post-pack
    unsigned*       vkey   = (unsigned*)alloc(NX * 4);
    unsigned*       vidx   = (unsigned*)alloc(NX * 4);
    _Float16*       xt     = (_Float16*)alloc(NX * 2);    // f16 transposed activations
    _Float16*       wqkvh  = (_Float16*)alloc((size_t)C5_ * C_ * 2);
    _Float16*       wouth  = (_Float16*)alloc((size_t)C_ * C_ * 2);

    // aliases into qkv (dead after dwconv): 320 MB available
    char* r1 = (char*)qkv;
    _Float16* Qp    = (_Float16*)(r1);
    _Float16* Kp    = (_Float16*)(r1 + (64ull  << 20));
    _Float16* Vp    = (_Float16*)(r1 + (128ull << 20));
    _Float16* attnp = (_Float16*)(r1 + (192ull << 20));
    float*    S     = (float*)   (r1 + (200ull << 20));
    float*    rnq   = (float*)   (r1 + (208ull << 20));
    float*    rnk   = (float*)   (r1 + (212ull << 20));

    // aliases into qkv_dw (dead after k_pack): 320 MB available
    char* r2 = (char*)qkv_dw;
    float*    o1    = (float*)(r2);
    float*    o2    = (float*)(r2 + (64ull  << 20));
    _Float16* prodt = (_Float16*)(r2 + (128ull << 20));   // 32 MB
    float*    cv    = (float*)(r2 + (192ull << 20));

    const int TPB = 256;

    // 0) pack weights to f16
    k_packw<<<(C5_ * C_ + TPB - 1) / TPB, TPB, 0, stream>>>(w_qkv, wqkvh, C5_ * C_);
    k_packw<<<(C_ * C_ + TPB - 1) / TPB, TPB, 0, stream>>>(w_out, wouth, C_ * C_);
    // 1) copy x -> x_work
    k_copy<<<(unsigned)((NX + TPB - 1) / TPB), TPB, 0, stream>>>(x, x_work, NX);
    // 2-3) stable sorts of first-half channels along H then W
    k_sort_h<<<dim3(W_, HALF_, B_), 256, 0, stream>>>(x_work, idxh);
    k_sort_w<<<dim3(H_, HALF_, B_), 256, 0, stream>>>(x_work, idxw);
    // 4) transpose+convert activations, then qkv = conv1x1 via WMMA (M=320,K=64)
    k_transpose_f16<<<dim3(HW_ / 64, B_), 256, 0, stream>>>(x_work, xt);
    k_conv1x1h<<<dim3(HW_ / 64, C5_ / 16, B_), 128, 0, stream>>>(wqkvh, xt, qkv, C5_);
    // 5) depthwise 3x3
    k_dwconv<<<(unsigned)((NQKV + TPB - 1) / TPB), TPB, 0, stream>>>(qkv, w_dw, qkv_dw, NQKV);
    // 6) sortable keys for v
    k_makekeys<<<(unsigned)((NX + TPB - 1) / TPB), TPB, 0, stream>>>(qkv_dw, vkey, vidx, NX);
    // 7) global bitonic argsort of each 65536-element channel (136 passes)
    for (int k = 2; k <= 65536; k <<= 1)
        for (int j = k >> 1; j >= 1; j >>= 1)
            k_bitonic<<<(unsigned)((NX + TPB - 1) / TPB), TPB, 0, stream>>>(vkey, vidx, j, k, NX);
    // 8) pack gathered Q/K and V^T tiles (f16); qkv buffer is reused here
    k_pack<<<(unsigned)((NPACK + TPB - 1) / TPB), TPB, 0, stream>>>(
        qkv_dw, vkey, vidx, Qp, Kp, Vp, NPACK);
    // 9) reciprocal row norms (normalization folded into logits)
    k_rownorm<<<dim3(NMAT_ * 64, 2), 256, 0, stream>>>(Qp, Kp, rnq, rnk);
    // 10) S = Q K^T via WMMA with LDS-staged (async) K tiles, K-split + atomics
    k_zero<<<(unsigned)((NMAT_ * 4096 + TPB - 1) / TPB), TPB, 0, stream>>>(S, (size_t)NMAT_ * 4096);
    k_qkT<<<dim3(8, NMAT_), 128, 0, stream>>>(Qp, Kp, S);
    // 11) scaled softmax_1 -> f16 attn
    k_softmax1<<<NMAT_ * 64, 64, 0, stream>>>(S, rnq, rnk, temp, attnp);
    // 12) out = attn @ V via WMMA, un-reshaped store into sorted-n space
    k_av<<<dim3(512, NMAT_), 128, 0, stream>>>(attnp, Vp, o1, o2);
    // 13) out1*out2 scattered back to original n, written f16-transposed
    k_mulscatter<<<(unsigned)((NX + TPB - 1) / TPB), TPB, 0, stream>>>(o1, o2, vidx, prodt, NX);
    // 14) final 1x1 conv (M=64,K=64) via WMMA
    k_conv1x1h<<<dim3(HW_ / 64, C_ / 16, B_), 128, 0, stream>>>(wouth, prodt, cv, C_);
    // 15) undo H/W sorts on first half channels, write d_out
    k_final<<<(unsigned)((NX + TPB - 1) / TPB), TPB, 0, stream>>>(
        cv, idxh, idxw, (float*)d_out, NX);
}
